// CrossLayerTranscoder_50156628082868
// MI455X (gfx1250) — matrix operand order repaired and checked by reading, not compiled, not run on hardware
//
#include <hip/hip_runtime.h>
#include <hip/hip_bf16.h>

// Cross-layer transcoder for MI455X (gfx1250, wave32, WMMA).
//   encode: feats[l] = relu(x[l] @ W_enc[l]^T + b_enc[l])   (1024x768)@(768x4096)
//   decode: recon[j] = sum_{i<=j} feats[i] @ W_dec[i,j] + b_dec[j]
// fp32 operands are converted to bf16 while staging into double-buffered LDS;
// matrix math is v_wmma_f32_16x16x32_bf16 with fp32 accumulators.
// Decode's A operand (bf16 feats) is streamed into LDS with
// global_load_async_to_lds_b64 (ASYNCcnt) to overlap with WMMA.

#define NL   8
#define NPOS 1024
#define DM   768
#define DTC  4096

#define LDS_STRIDE 18   // dwords per 32-wide K row (16 data + 2 pad; keeps b64 stores 8B-aligned,
                        // and 18*m mod 64 is distinct for m=0..15 -> conflict-free frag reads)
#define USE_ASYNC_LDS 1

#if defined(__has_builtin)
#  if __has_builtin(__builtin_amdgcn_cvt_pk_bf16_f32)
#    define HAVE_CVT_PK_BF16 1
#  endif
#endif
#ifndef HAVE_CVT_PK_BF16
#  define HAVE_CVT_PK_BF16 0
#endif

typedef __bf16 v16bf __attribute__((ext_vector_type(16)));
typedef float  v8f   __attribute__((ext_vector_type(8)));

union Frag { unsigned int u[8]; v16bf v; };

// Pack two fp32 -> two bf16 (RNE) in one dword.
__device__ __forceinline__ unsigned int packf2(float x, float y) {
#if HAVE_CVT_PK_BF16
  auto h = __builtin_amdgcn_cvt_pk_bf16_f32(x, y);   // single v_cvt_pk_bf16_f32
  unsigned int u;
  __builtin_memcpy(&u, &h, sizeof(u));
  return u;
#else
  __hip_bfloat162 h = __float22bfloat162_rn(make_float2(x, y));
  unsigned int u;
  __builtin_memcpy(&u, &h, sizeof(u));
  return u;
#endif
}

// Scalar fp32 -> bf16 (RNE), manual (header lacks __float2bfloat16_rn here).
__device__ __forceinline__ unsigned short f2bf(float a) {
  unsigned int u = __float_as_uint(a);
  u += 0x7FFFu + ((u >> 16) & 1u);
  return (unsigned short)(u >> 16);
}

__device__ __forceinline__ void async_wait0() {
#if USE_ASYNC_LDS
  asm volatile("s_wait_asynccnt 0x0" ::: "memory");
#endif
}

#if USE_ASYNC_LDS
// LDS[lds_off .. +7] = MEM[gaddr .. +7], tracked by ASYNCcnt.
__device__ __forceinline__ void async_load_b64(unsigned int lds_off, const void* gaddr) {
  asm volatile("global_load_async_to_lds_b64 %0, %1, off"
               :: "v"(lds_off), "v"((unsigned long long)gaddr) : "memory");
}
#endif

// ---------------------------------------------------------------------------
// WMMA inner step on one 128x128 tile, K-slab of 32. LDS holds packed bf16
// pairs (uint = {K, K+1}).
//   lA[row 0..127][kp 0..15] : A rows (M-major), K-pairs contiguous
//   lB[col 0..127][kp 0..15] : B cols (N-major), K-pairs contiguous
// A 16-bit 16x32 layout: lane m=lane&15, half=lane>>4;
//   VGPR v in 0..7: K = half*16 + 2v (+1), same k order as B (not the table order).
// B 16-bit 32x16 layout: lane n=lane&15, half=lane>>4; VGPR v: K = half*16 + 2v (+1) (this code's order, not the table order).
// ---------------------------------------------------------------------------
__device__ __forceinline__ void wmma_tile_step(
    const unsigned int (*lA)[LDS_STRIDE], const unsigned int (*lB)[LDS_STRIDE],
    int wr, int wc, int lane, v8f acc[4][2])
{
  const int lm   = lane & 15;
  const int half = lane >> 4;

  Frag bfr[2];
#pragma unroll
  for (int nt = 0; nt < 2; ++nt) {
    const int n = wc * 32 + nt * 16 + lm;
#pragma unroll
    for (int v = 0; v < 8; ++v) bfr[nt].u[v] = lB[n][half * 8 + v];
  }

#pragma unroll
  for (int mt = 0; mt < 4; ++mt) {
    Frag afr;
    const int m = wr * 64 + mt * 16 + lm;
#pragma unroll
    for (int v = 0; v < 4; ++v) afr.u[v]     = lA[m][half * 8 + v];
#pragma unroll
    for (int v = 0; v < 4; ++v) afr.u[4 + v] = lA[m][half * 8 + 4 + v];
#pragma unroll
    for (int nt = 0; nt < 2; ++nt) {
      acc[mt][nt] = __builtin_amdgcn_wmma_f32_16x16x32_bf16(
          false, afr.v, false, bfr[nt].v, (short)0, acc[mt][nt], false, false);
    }
  }
}

// ---------------------------------------------------------------------------
// Encode: feats[l,b,f] = relu(sum_d x[l,b,d] * W_enc[l,f,d] + b_enc[l,f])
// Both operands K(=d)-contiguous: float4 global loads, pack, b64 LDS stores.
// Grid: (DTC/128, NPOS/128, NL), block 256 (8 waves). Double-buffered LDS.
// ---------------------------------------------------------------------------
__global__ __launch_bounds__(256) void clt_encode_kernel(
    const float* __restrict__ x, const float* __restrict__ Wenc,
    const float* __restrict__ benc, unsigned short* __restrict__ feats)
{
  __shared__ unsigned int lA[2][128][LDS_STRIDE];
  __shared__ unsigned int lB[2][128][LDS_STRIDE];

  const int tid  = threadIdx.x;
  const int lane = tid & 31;
  const int wave = tid >> 5;
  const int wr   = wave & 1;   // 2 wave-rows of 64
  const int wc   = wave >> 1;  // 4 wave-cols of 32
  const int l       = blockIdx.z;
  const int rowBase = blockIdx.y * 128;  // positions
  const int colBase = blockIdx.x * 128;  // features

  const float* __restrict__ xA = x    + (size_t)l * NPOS * DM;
  const float* __restrict__ wB = Wenc + (size_t)l * DTC  * DM;

  // Per-thread staging slots: 1024 float4 units per operand tile, 4 per thread.
  // unit u: row = u>>3 (0..127), q = u&7 (float4 within the 32-wide K).
  const int srow[4] = { (tid + 0*256) >> 3, (tid + 1*256) >> 3,
                        (tid + 2*256) >> 3, (tid + 3*256) >> 3 };
  const int sq = tid & 7;

  v8f acc[4][2];
#pragma unroll
  for (int mt = 0; mt < 4; ++mt)
#pragma unroll
    for (int nt = 0; nt < 2; ++nt)
#pragma unroll
      for (int r = 0; r < 8; ++r) acc[mt][nt][r] = 0.0f;

  const int KT = DM / 32;  // 24

  // Prologue: stage slab 0 into buffer 0.
#pragma unroll
  for (int s = 0; s < 4; ++s) {
    const float4 fa = *(const float4*)(xA + (size_t)(rowBase + srow[s]) * DM + sq * 4);
    const float4 fb = *(const float4*)(wB + (size_t)(colBase + srow[s]) * DM + sq * 4);
    uint2 pa = { packf2(fa.x, fa.y), packf2(fa.z, fa.w) };
    uint2 pb = { packf2(fb.x, fb.y), packf2(fb.z, fb.w) };
    *(uint2*)&lA[0][srow[s]][sq * 2] = pa;
    *(uint2*)&lB[0][srow[s]][sq * 2] = pb;
  }
  __syncthreads();

  int p = 0;
  for (int kb = 0; kb < KT; ++kb) {
    const bool next = (kb + 1 < KT);
    float4 ra[4], rb[4];
    if (next) {
#pragma unroll
      for (int s = 0; s < 4; ++s) {
        ra[s] = *(const float4*)(xA + (size_t)(rowBase + srow[s]) * DM + (kb + 1) * 32 + sq * 4);
        rb[s] = *(const float4*)(wB + (size_t)(colBase + srow[s]) * DM + (kb + 1) * 32 + sq * 4);
      }
      if (kb + 2 < KT) {  // prefetch the slab after next for the weight stream
        const float* pf = wB + (size_t)(colBase + (tid >> 1)) * DM + (kb + 2) * 32 + (tid & 1) * 16;
        __builtin_prefetch(pf, 0, 1);
      }
    }

    wmma_tile_step(lA[p], lB[p], wr, wc, lane, acc);

    if (next) {
#pragma unroll
      for (int s = 0; s < 4; ++s) {
        uint2 pa = { packf2(ra[s].x, ra[s].y), packf2(ra[s].z, ra[s].w) };
        uint2 pb = { packf2(rb[s].x, rb[s].y), packf2(rb[s].z, rb[s].w) };
        *(uint2*)&lA[p ^ 1][srow[s]][sq * 2] = pa;
        *(uint2*)&lB[p ^ 1][srow[s]][sq * 2] = pb;
      }
    }
    __syncthreads();
    p ^= 1;
  }

  // Epilogue: bias + ReLU, store bf16 feats (decode A-operand).
  const int lm   = lane & 15;
  const int half = lane >> 4;
  unsigned short* __restrict__ fOut = feats + (size_t)l * NPOS * DTC;
#pragma unroll
  for (int nt = 0; nt < 2; ++nt) {
    const int f    = colBase + wc * 32 + nt * 16 + lm;
    const float bv = benc[l * DTC + f];
#pragma unroll
    for (int mt = 0; mt < 4; ++mt) {
#pragma unroll
      for (int r = 0; r < 8; ++r) {
        const int m = rowBase + wr * 64 + mt * 16 + half * 8 + r;
        float v = acc[mt][nt][r] + bv;
        v = v > 0.0f ? v : 0.0f;
        fOut[(size_t)m * DTC + f] = f2bf(v);
      }
    }
  }
}

// ---------------------------------------------------------------------------
// Decode: recon[j,b,d] = sum_{i<=j} sum_f feats[i,b,f] * W_dec[i,j,f,d] + b_dec[j,d]
// A = bf16 feats (K=f contiguous) -> async-copied straight into LDS.
// B = W_dec[i,j] is (f,d) d-contiguous -> transpose-stage via float4 row pairs.
// Grid: (DM/128, NPOS/128, NL), block 256. Double-buffered LDS.
// ---------------------------------------------------------------------------
__global__ __launch_bounds__(256) void clt_decode_kernel(
    const unsigned short* __restrict__ feats, const float* __restrict__ Wdec,
    const float* __restrict__ bdec, float* __restrict__ out)
{
  __shared__ unsigned int lA[2][128][LDS_STRIDE];
  __shared__ unsigned int lB[2][128][LDS_STRIDE];

  const int tid  = threadIdx.x;
  const int lane = tid & 31;
  const int wave = tid >> 5;
  const int wr   = wave & 1;
  const int wc   = wave >> 1;
  const int j       = blockIdx.z;
  const int rowBase = blockIdx.y * 128;  // positions
  const int colBase = blockIdx.x * 128;  // d_model (6 tiles)

  v8f acc[4][2];
#pragma unroll
  for (int mt = 0; mt < 4; ++mt)
#pragma unroll
    for (int nt = 0; nt < 2; ++nt)
#pragma unroll
      for (int r = 0; r < 8; ++r) acc[mt][nt][r] = 0.0f;

  // A staging: 1024 8-byte chunks per tile, 4 per thread.
  // chunk u: row = u>>3 (0..127), ch = u&7 (8B chunk within the 64B row).
  const int arow[4] = { (tid + 0*256) >> 3, (tid + 1*256) >> 3,
                        (tid + 2*256) >> 3, (tid + 3*256) >> 3 };
  const int ach = tid & 7;
  // B staging: 512 float4-pair units per tile, 2 per thread.
  // unit u: d4 = (u&31)*4, kp = u>>5 (0..15).
  const int bd4[2] = { ((tid + 0*256) & 31) * 4, ((tid + 1*256) & 31) * 4 };
  const int bkp[2] = { (tid + 0*256) >> 5,       (tid + 1*256) >> 5 };

  const int T = (j + 1) * (DTC / 32);  // flattened (i, kb) iteration count

  // Prologue: stage iteration 0 into buffer 0.
  {
    const unsigned short* fA = feats;
    const float* wD = Wdec + ((size_t)0 * NL + j) * (size_t)DTC * DM;
#if USE_ASYNC_LDS
#pragma unroll
    for (int s = 0; s < 4; ++s) {
      const void* g = fA + (size_t)(rowBase + arow[s]) * DTC + ach * 4;
      async_load_b64((unsigned int)(unsigned long long)&lA[0][arow[s]][ach * 2], g);
    }
#else
#pragma unroll
    for (int s = 0; s < 4; ++s) {
      uint2 v = *(const uint2*)(fA + (size_t)(rowBase + arow[s]) * DTC + ach * 4);
      *(uint2*)&lA[0][arow[s]][ach * 2] = v;
    }
#endif
#pragma unroll
    for (int s = 0; s < 2; ++s) {
      const float* r0 = wD + (size_t)(2 * bkp[s]) * DM + colBase + bd4[s];
      const float4 f0 = *(const float4*)r0;
      const float4 f1 = *(const float4*)(r0 + DM);
      lB[0][bd4[s] + 0][bkp[s]] = packf2(f0.x, f1.x);
      lB[0][bd4[s] + 1][bkp[s]] = packf2(f0.y, f1.y);
      lB[0][bd4[s] + 2][bkp[s]] = packf2(f0.z, f1.z);
      lB[0][bd4[s] + 3][bkp[s]] = packf2(f0.w, f1.w);
    }
  }
  async_wait0();
  __syncthreads();

  int p = 0;
  for (int it = 0; it < T; ++it) {
    const int  nit  = it + 1;
    const bool next = (nit < T);
    float4 f0[2], f1[2];
    if (next) {
      const int ni  = nit >> 7;       // DTC/32 == 128 slabs per source layer
      const int nkb = nit & 127;
      const unsigned short* fA = feats + (size_t)ni * NPOS * DTC;
      const float* wD = Wdec + ((size_t)ni * NL + j) * (size_t)DTC * DM;
#if USE_ASYNC_LDS
#pragma unroll
      for (int s = 0; s < 4; ++s) {
        const void* g = fA + (size_t)(rowBase + arow[s]) * DTC + nkb * 32 + ach * 4;
        async_load_b64((unsigned int)(unsigned long long)&lA[p ^ 1][arow[s]][ach * 2], g);
      }
#else
#pragma unroll
      for (int s = 0; s < 4; ++s) {
        uint2 v = *(const uint2*)(fA + (size_t)(rowBase + arow[s]) * DTC + nkb * 32 + ach * 4);
        *(uint2*)&lA[p ^ 1][arow[s]][ach * 2] = v;
      }
#endif
#pragma unroll
      for (int s = 0; s < 2; ++s) {
        const float* r0 = wD + (size_t)(nkb * 32 + 2 * bkp[s]) * DM + colBase + bd4[s];
        f0[s] = *(const float4*)r0;
        f1[s] = *(const float4*)(r0 + DM);
      }
      if (nkb + 1 < 128) {  // prefetch the weight slab after next
        const float* pf = wD + (size_t)((nkb + 1) * 32 + (tid >> 3)) * DM + colBase + (tid & 7) * 16;
        __builtin_prefetch(pf, 0, 1);
      }
    }

    wmma_tile_step(lA[p], lB[p], wr, wc, lane, acc);

    if (next) {
#pragma unroll
      for (int s = 0; s < 2; ++s) {
        lB[p ^ 1][bd4[s] + 0][bkp[s]] = packf2(f0[s].x, f1[s].x);
        lB[p ^ 1][bd4[s] + 1][bkp[s]] = packf2(f0[s].y, f1[s].y);
        lB[p ^ 1][bd4[s] + 2][bkp[s]] = packf2(f0[s].z, f1[s].z);
        lB[p ^ 1][bd4[s] + 3][bkp[s]] = packf2(f0[s].w, f1[s].w);
      }
    }
    async_wait0();
    __syncthreads();
    p ^= 1;
  }

  // Epilogue: + b_dec, fp32 store.
  const int lm   = lane & 15;
  const int half = lane >> 4;
  float* __restrict__ oJ = out + (size_t)j * NPOS * DM;
#pragma unroll
  for (int nt = 0; nt < 2; ++nt) {
    const int d    = colBase + wc * 32 + nt * 16 + lm;
    const float bv = bdec[j * DM + d];
#pragma unroll
    for (int mt = 0; mt < 4; ++mt) {
#pragma unroll
      for (int r = 0; r < 8; ++r) {
        const int m = rowBase + wr * 64 + mt * 16 + half * 8 + r;
        oJ[(size_t)m * DM + d] = acc[mt][nt][r] + bv;
      }
    }
  }
}

extern "C" void kernel_launch(void* const* d_in, const int* in_sizes, int n_in,
                              void* d_out, int out_size, void* d_ws, size_t ws_size,
                              hipStream_t stream) {
  const float* x    = (const float*)d_in[0];  // (8,1024,768)
  const float* Wenc = (const float*)d_in[1];  // (8,4096,768)
  const float* benc = (const float*)d_in[2];  // (8,4096)
  const float* Wdec = (const float*)d_in[3];  // (8,8,4096,768)
  const float* bdec = (const float*)d_in[4];  // (8,768)
  float* out = (float*)d_out;                 // (8,1024,768)

  // bf16 feats intermediate in workspace: 8*1024*4096*2 = 64 MB (fits in L2).
  unsigned short* feats = (unsigned short*)d_ws;

  dim3 blk(256, 1, 1);
  dim3 egrid(DTC / 128, NPOS / 128, NL);  // 32 x 8 x 8
  clt_encode_kernel<<<egrid, blk, 0, stream>>>(x, Wenc, benc, feats);

  dim3 dgrid(DM / 128, NPOS / 128, NL);   // 6 x 8 x 8
  clt_decode_kernel<<<dgrid, blk, 0, stream>>>(feats, Wdec, bdec, out);
}
